// SimpleLLM_77738908058256
// MI455X (gfx1250) — matrix operand order repaired and checked by reading, not compiled, not run on hardware
//
#include <hip/hip_runtime.h>
#include <cmath>

typedef __attribute__((ext_vector_type(8)))  __bf16 v8bf;
typedef __attribute__((ext_vector_type(16))) __bf16 v16bf;
typedef __attribute__((ext_vector_type(8)))  float  v8f;

constexpr int Vv = 32000, Hh = 2048, NH = 16, NKV = 4, HD = 128, FF = 8192, LL = 2;
constexpr int Bb = 2, Ss = 1024, MM = Bb * Ss;   // 2048 token rows

// ---------------------------------------------------------------------------
// TDM (Tensor Data Mover) support — this toolchain exposes the 6-arg builtin:
//   (uint32x4 g0, int32x8 g1, int32x4 g2, int32x4 g3, int32x8 g4, i32 cpol)
// Guarded so the host pass falls back to synchronous staging.
// ---------------------------------------------------------------------------
#if defined(__has_builtin)
#if __has_builtin(__builtin_amdgcn_tensor_load_to_lds) && \
    __has_builtin(__builtin_amdgcn_s_wait_tensorcnt)
#define USE_TDM 1
#endif
#endif
#ifndef USE_TDM
#define USE_TDM 0
#endif

#if USE_TDM
typedef __attribute__((ext_vector_type(4))) unsigned int u32x4;
typedef __attribute__((ext_vector_type(8))) int          i32x8;
typedef __attribute__((ext_vector_type(4))) int          i32x4;

// LDS aperture: flat = {SHARED_BASE[63:32], lds_offset[31:0]} -> low 32 bits
// of a generic pointer into __shared__ are the LDS byte offset (ISA 10.2).
__device__ inline unsigned lds_offset_of(const void* p) {
  return (unsigned)(size_t)p;
}

// 2D bf16 tile DMA: 128 x 32 elements of an nRows-row tensor, row stride rowStride (elements),
// LDS padding 4 DWORDs after every 16 DWORDs -> 40-element LDS row stride.
__device__ inline void tdm_load_tile_bf16(const __bf16* gsrc, unsigned ldsOff,
                                          int rowElems, int nRows,
                                          int rowStride) {
  unsigned long long ga = (unsigned long long)gsrc;
  u32x4 g0;
  g0[0] = 1u;                                  // D# count = 1 (user mode)
  g0[1] = ldsOff;                              // lds_addr
  g0[2] = (unsigned)ga;                        // global_addr[31:0]
  g0[3] = (unsigned)(ga >> 32) | (2u << 30);   // global_addr[56:32] | type=2
  i32x8 g1;
  g1[0] = (int)((1u << 16) |                   // data_size = 2 bytes
                (1u << 20) |                   // pad_enable
                (3u << 22) |                   // pad_interval: 16 DWORDs
                (3u << 25));                   // pad_amount: 4 DWORDs
  g1[1] = (int)(((unsigned)rowElems & 0xFFFFu) << 16);          // dim0 lo16
  g1[2] = (int)((((unsigned)rowElems >> 16) & 0xFFFFu) |
                (((unsigned)nRows & 0xFFFFu) << 16));           // dim0 hi|dim1 lo
  g1[3] = (int)((((unsigned)nRows >> 16) & 0xFFFFu) |
                (32u << 16));                                   // dim1 hi|tile_dim0
  g1[4] = (int)128u;                                            // tile_dim1=128
  g1[5] = (int)rowStride;                                       // dim0_stride lo
  g1[6] = 0;
  g1[7] = 0;
  i32x4 gz4 = {0, 0, 0, 0};                    // groups 2/3 unused (2D tensor)
  i32x8 gz8 = {0, 0, 0, 0, 0, 0, 0, 0};        // trailing group (unused)
  __builtin_amdgcn_tensor_load_to_lds(g0, g1, gz4, gz4, gz8, 0);
}
#endif

// ---------------------------------------------------------------------------
// WMMA helpers (CDNA5: wave32, 16x16x32 bf16 -> f32)
// ---------------------------------------------------------------------------
__device__ inline v16bf make_frag(const __bf16* p0, const __bf16* p1) {
  v8bf lo = *(const v8bf*)p0;
  v8bf hi = *(const v8bf*)p1;
  v16bf r;
#pragma unroll
  for (int i = 0; i < 8; ++i) { r[i] = lo[i]; r[i + 8] = hi[i]; }
  return r;
}

__device__ inline v8f wmma_bf16(v16bf a, v16bf b, v8f c) {
  return __builtin_amdgcn_wmma_f32_16x16x32_bf16(
      /*neg_a=*/false, a, /*neg_b=*/false, b,
      /*c_mod=*/(short)0, c, /*reuse_a=*/false, /*reuse_b=*/false);
}

// ---------------------------------------------------------------------------
// GEMM: Y[M,N] = A(bf16)[M,K] @ W(f32,[N,K])^T.
// Block 256 thr = 8 waves; tile 128x128x32; wave tile 64x32 = 4x2 wmma tiles.
// A tiles: double-buffered TDM DMA into LDS (hardware padded rows),
//          synchronized with s_wait_tensorcnt.
// W tiles: register-pipelined f32 loads converted to bf16 into LDS.
// EPI: 0 = store bf16, 1 = store f32, 2 = add into f32 (residual)
// ---------------------------------------------------------------------------
constexpr int BM = 128, BN = 128, BK = 32, BKP = 40;

template <int EPI>
__global__ __launch_bounds__(256) void gemm_xw_kernel(
    const __bf16* __restrict__ A, const float* __restrict__ W,
    void* __restrict__ Y, int M, int N, int K) {
  __shared__ __bf16 As[2][BM][BKP];
  __shared__ __bf16 Bs[2][BN][BKP];
  const int tid  = threadIdx.x;
  const int lane = tid & 31, wv = tid >> 5;
  const int wm = wv & 1, wn = wv >> 1;            // waves: 2 (M) x 4 (N)
  const int bm = blockIdx.y * BM, bn = blockIdx.x * BN;
  const int msel = lane & 15, ksel = lane >> 4;   // wmma lane decomposition
  const int lr = tid >> 1, lc = (tid & 1) * 16;   // staging: 128 rows x 2 chunks
  const int nk = K / BK;

  v8f vzero = {};
  v8f acc[4][2];
#pragma unroll
  for (int i = 0; i < 4; ++i)
#pragma unroll
    for (int j = 0; j < 2; ++j) acc[i][j] = vzero;

  const float* wGlob = W + (size_t)(bn + lr) * K + lc;
#if !USE_TDM
  const __bf16* aRow = A + (size_t)(bm + lr) * K + lc;
#endif

#if USE_TDM
  const __bf16* aBase = A + (size_t)bm * K;       // block-uniform tile base
  unsigned asOff[2] = {lds_offset_of(&As[0][0][0]),
                       lds_offset_of(&As[1][0][0])};
  if (wv == 0)  // one DMA per tile; TDM ignores EXEC, tracked by TENSORcnt
    tdm_load_tile_bf16(aBase, asOff[0], K, M - bm, K);
#endif

  // register-prefetch first W tile
  float4 wreg[4];
#pragma unroll
  for (int j = 0; j < 4; ++j) wreg[j] = ((const float4*)wGlob)[j];

  for (int i = 0; i < nk; ++i) {
    const int cur = i & 1;
    const int k0 = i * BK;

#if USE_TDM
    if (i + 1 < nk && wv == 0)  // kick DMA for next A tile into other buffer
      tdm_load_tile_bf16(aBase + k0 + BK, asOff[cur ^ 1], K - k0 - BK,
                         M - bm, K);
#else
    const __bf16* ap = aRow + k0;
    *(v8bf*)&As[cur][lr][lc]     = *(const v8bf*)ap;
    *(v8bf*)&As[cur][lr][lc + 8] = *(const v8bf*)(ap + 8);
#endif

    // stage current W tile (f32 -> bf16), then prefetch next into registers
#pragma unroll
    for (int j = 0; j < 4; ++j) {
      __bf16* d = &Bs[cur][lr][lc + j * 4];
      d[0] = (__bf16)wreg[j].x; d[1] = (__bf16)wreg[j].y;
      d[2] = (__bf16)wreg[j].z; d[3] = (__bf16)wreg[j].w;
    }
    if (i + 1 < nk) {
      const float* wp = wGlob + k0 + BK;
#pragma unroll
      for (int j = 0; j < 4; ++j) wreg[j] = ((const float4*)wp)[j];
      __builtin_prefetch(wp + BK, 0, 0);  // global_prefetch_b8 for k+2
    }

#if USE_TDM
    if (wv == 0) {
      if (i + 1 < nk) __builtin_amdgcn_s_wait_tensorcnt(1);  // older tile done
      else            __builtin_amdgcn_s_wait_tensorcnt(0);
    }
#endif
    __syncthreads();

    // A frag (16x32): lane m = msel, K chunks ksel*8 and ksel*8+16
    v16bf afr[4], bfr[2];
#pragma unroll
    for (int mi = 0; mi < 4; ++mi) {
      const __bf16* p = &As[cur][wm * 64 + mi * 16 + msel][ksel * 8];
      afr[mi] = make_frag(p, p + 16);
    }
    // B frag (32x16): lane n = msel, K chunks ksel*8 and ksel*8+16
#pragma unroll
    for (int ni = 0; ni < 2; ++ni) {
      const __bf16* p = &Bs[cur][wn * 32 + ni * 16 + msel][ksel * 8];
      bfr[ni] = make_frag(p, p + 16);
    }
#pragma unroll
    for (int mi = 0; mi < 4; ++mi)
#pragma unroll
      for (int ni = 0; ni < 2; ++ni)
        acc[mi][ni] = wmma_bf16(afr[mi], bfr[ni], acc[mi][ni]);
    __syncthreads();
  }

  // epilogue: C layout = VGPR r, M = r + 8*(lane>=16), N = lane&15
#pragma unroll
  for (int mi = 0; mi < 4; ++mi)
#pragma unroll
    for (int ni = 0; ni < 2; ++ni)
#pragma unroll
      for (int r = 0; r < 8; ++r) {
        int gm = bm + wm * 64 + mi * 16 + r + 8 * ksel;
        int gn = bn + wn * 32 + ni * 16 + msel;
        float v = acc[mi][ni][r];
        size_t o = (size_t)gm * N + gn;
        if (EPI == 0)      ((__bf16*)Y)[o] = (__bf16)v;
        else if (EPI == 1) ((float*)Y)[o]  = v;
        else               ((float*)Y)[o] += v;
      }
}

// ---------------------------------------------------------------------------
// Attention: block = (b*head, 64-query tile), 4 waves x 16 query rows.
// Scores for 64 rows x <=1024 keys live in dynamic LDS (264KB of 320KB/WGP).
// ---------------------------------------------------------------------------
__global__ __launch_bounds__(128) void attn_kernel(
    const __bf16* __restrict__ Q, const __bf16* __restrict__ Kc,
    const __bf16* __restrict__ Vt, __bf16* __restrict__ Ctx) {
  extern __shared__ float P[];                 // [64][SROW]
  constexpr int SROW = 1032;
  const int bh = blockIdx.x, qb = blockIdx.y;
  const int b = bh / NH, head = bh % NH, kv = head % NKV;
  const int tid = threadIdx.x, lane = tid & 31, wv = tid >> 5;
  const int msel = lane & 15, ksel = lane >> 4;
  const int nKeys = (qb + 1) * 64;             // causal: skip future key tiles
  const float scale = 0.08838834764831845f;    // 1/sqrt(128)

  // Preload Q fragments (A matrix, 4 chunks over HD=128)
  const int qrow = qb * 64 + wv * 16 + msel;
  const __bf16* qp = Q + ((size_t)(b * Ss + qrow) * NH + head) * HD;
  v16bf qa[4];
#pragma unroll
  for (int c = 0; c < 4; ++c) {
    const __bf16* p = qp + c * 32 + ksel * 16;
    qa[c] = make_frag(p, p + 8);
  }
  v8f vzero = {};

  // Phase 1: S = Q @ K^T, scaled + causal mask, into LDS
  for (int n0 = 0; n0 < nKeys; n0 += 16) {
    const __bf16* kp =
        Kc + ((size_t)(b * Ss + n0 + msel) * NKV + kv) * HD + ksel * 16;
    v8f sc = vzero;
#pragma unroll
    for (int c = 0; c < 4; ++c) {
      v16bf kf = *(const v16bf*)(kp + c * 32);   // 32B aligned contiguous K-dim
      sc = wmma_bf16(qa[c], kf, sc);
    }
#pragma unroll
    for (int r = 0; r < 8; ++r) {
      int m = r + 8 * ksel;
      int qg = qb * 64 + wv * 16 + m;
      int key = n0 + msel;
      float sv = sc[r] * scale;
      if (key > qg) sv = -INFINITY;
      P[(wv * 16 + m) * SROW + key] = sv;
    }
  }
  __syncthreads();

  // Phase 2: softmax, 2 lanes per row, combine with shfl_xor
  {
    int row = wv * 16 + (lane >> 1);
    int half = lane & 1;
    float* pr = P + row * SROW;
    int cnt = nKeys >> 1, st = half * cnt;
    float mx = -INFINITY;
    for (int i = 0; i < cnt; ++i) mx = fmaxf(mx, pr[st + i]);
    mx = fmaxf(mx, __shfl_xor(mx, 1, 32));
    float sum = 0.f;
    for (int i = 0; i < cnt; ++i) {
      float e = __expf(pr[st + i] - mx);
      pr[st + i] = e;
      sum += e;
    }
    sum += __shfl_xor(sum, 1, 32);
    float rs = 1.f / sum;
    for (int i = 0; i < cnt; ++i) pr[st + i] *= rs;
  }
  __syncthreads();

  // Phase 3: ctx = P @ V  (V pre-transposed to [b,kv,hd,s])
  v8f acc[8];
#pragma unroll
  for (int n = 0; n < 8; ++n) acc[n] = vzero;
  const float* prow = P + (wv * 16 + msel) * SROW;
  for (int k0 = 0; k0 < nKeys; k0 += 32) {
    v16bf pa;
#pragma unroll
    for (int j = 0; j < 8; ++j) {
      pa[j]     = (__bf16)prow[k0 + ksel * 16 + j];
      pa[j + 8] = (__bf16)prow[k0 + ksel * 16 + 8 + j];
    }
#pragma unroll
    for (int n = 0; n < 8; ++n) {
      const __bf16* vp =
          Vt + ((size_t)((b * NKV + kv) * HD) + n * 16 + msel) * Ss + k0 + ksel * 16;
      v16bf vf = *(const v16bf*)vp;
      acc[n] = wmma_bf16(pa, vf, acc[n]);
    }
  }
#pragma unroll
  for (int n = 0; n < 8; ++n)
#pragma unroll
    for (int r = 0; r < 8; ++r) {
      int m = r + 8 * ksel;
      size_t tok = (size_t)b * Ss + qb * 64 + wv * 16 + m;
      int col = head * HD + n * 16 + msel;
      Ctx[tok * (NH * HD) + col] = (__bf16)acc[n][r];
    }
}

// ---------------------------------------------------------------------------
// Elementwise kernels
// ---------------------------------------------------------------------------
__global__ void embed_kernel(const int* __restrict__ ids,
                             const float* __restrict__ emb,
                             float* __restrict__ h) {
  int i = blockIdx.x * blockDim.x + threadIdx.x;   // MM*Hh = 4.2M
  if (i >= MM * Hh) return;
  int row = i / Hh, c = i % Hh;
  h[i] = emb[(size_t)ids[row] * Hh + c];
}

__global__ __launch_bounds__(256) void rmsnorm_kernel(
    const float* __restrict__ h, const float* __restrict__ w,
    __bf16* __restrict__ out) {
  int row = blockIdx.x;
  const float* x = h + (size_t)row * Hh;
  __shared__ float red[256];
  float s = 0.f;
  for (int i = threadIdx.x; i < Hh; i += 256) { float v = x[i]; s += v * v; }
  red[threadIdx.x] = s;
  __syncthreads();
  for (int off = 128; off; off >>= 1) {
    if (threadIdx.x < off) red[threadIdx.x] += red[threadIdx.x + off];
    __syncthreads();
  }
  float scale = rsqrtf(red[0] / (float)Hh + 1e-5f);
  __bf16* o = out + (size_t)row * Hh;
  for (int i = threadIdx.x; i < Hh; i += 256)
    o[i] = (__bf16)(x[i] * scale * w[i]);
}

__global__ void rope_kernel(__bf16* __restrict__ x, int nheads, int total) {
  int p = blockIdx.x * blockDim.x + threadIdx.x;
  if (p >= total) return;
  int i = p & 63;                       // pair index 0..HD/2-1
  int t = p >> 6;
  int head = t % nheads;
  int tok = t / nheads;
  int s = tok % Ss;
  float theta = __powf(10000.f, (-2.f * (float)i) / (float)HD);
  float ang = (float)s * theta;
  float c = __cosf(ang), sn = __sinf(ang);
  __bf16* px = x + ((size_t)tok * nheads + head) * HD + 2 * i;
  float x0 = (float)px[0], x1 = (float)px[1];
  px[0] = (__bf16)(x0 * c - x1 * sn);
  px[1] = (__bf16)(x1 * c + x0 * sn);
}

__global__ void vtrans_kernel(const __bf16* __restrict__ v,
                              __bf16* __restrict__ vt) {
  int i = blockIdx.x * blockDim.x + threadIdx.x;   // MM*NKV*HD = 1M
  if (i >= MM * NKV * HD) return;
  int hd = i % HD;
  int t = i / HD;
  int kvh = t % NKV;
  int t2 = t / NKV;
  int s = t2 % Ss;
  int b = t2 / Ss;
  vt[((size_t)((b * NKV + kvh) * HD) + hd) * Ss + s] = v[i];
}

__global__ void silu_mul_kernel(const __bf16* __restrict__ g,
                                const __bf16* __restrict__ u,
                                __bf16* __restrict__ o, int total) {
  int i = blockIdx.x * blockDim.x + threadIdx.x;
  if (i >= total) return;
  float gv = (float)g[i], uv = (float)u[i];
  o[i] = (__bf16)((gv / (1.f + __expf(-gv))) * uv);
}

// ---------------------------------------------------------------------------
// Launcher
// ---------------------------------------------------------------------------
extern "C" void kernel_launch(void* const* d_in, const int* in_sizes, int n_in,
                              void* d_out, int out_size, void* d_ws,
                              size_t ws_size, hipStream_t stream) {
  const int*   ids   = (const int*)d_in[0];
  const float* emb   = (const float*)d_in[1];
  const float* Wq    = (const float*)d_in[2];
  const float* Wk    = (const float*)d_in[3];
  const float* Wv    = (const float*)d_in[4];
  const float* Wo    = (const float*)d_in[5];
  const float* innrm = (const float*)d_in[6];
  const float* pnrm  = (const float*)d_in[7];
  const float* Wg    = (const float*)d_in[8];
  const float* Wu    = (const float*)d_in[9];
  const float* Wd    = (const float*)d_in[10];
  const float* fnrm  = (const float*)d_in[11];
  const float* lmh   = (const float*)d_in[12];
  float* out = (float*)d_out;

  char* w = (char*)d_ws;
  auto carve = [&](size_t bytes) -> char* {
    char* p = w;
    w += (bytes + 255) & ~(size_t)255;
    return p;
  };
  float*  h    = (float*) carve((size_t)MM * Hh * 4);
  __bf16* xb   = (__bf16*)carve((size_t)MM * Hh * 2);
  __bf16* qb   = (__bf16*)carve((size_t)MM * NH * HD * 2);
  __bf16* kb   = (__bf16*)carve((size_t)MM * NKV * HD * 2);
  __bf16* vb   = (__bf16*)carve((size_t)MM * NKV * HD * 2);
  __bf16* vtb  = (__bf16*)carve((size_t)MM * NKV * HD * 2);
  __bf16* ctxb = (__bf16*)carve((size_t)MM * NH * HD * 2);
  __bf16* gb   = (__bf16*)carve((size_t)MM * FF * 2);
  __bf16* ub   = (__bf16*)carve((size_t)MM * FF * 2);
  __bf16* fb   = (__bf16*)carve((size_t)MM * FF * 2);

  embed_kernel<<<(MM * Hh + 255) / 256, 256, 0, stream>>>(ids, emb, h);

  for (int l = 0; l < LL; ++l) {
    rmsnorm_kernel<<<MM, 256, 0, stream>>>(h, innrm + (size_t)l * Hh, xb);
    gemm_xw_kernel<0><<<dim3((NH * HD) / BN, MM / BM), 256, 0, stream>>>(
        xb, Wq + (size_t)l * NH * HD * Hh, qb, MM, NH * HD, Hh);
    gemm_xw_kernel<0><<<dim3((NKV * HD) / BN, MM / BM), 256, 0, stream>>>(
        xb, Wk + (size_t)l * NKV * HD * Hh, kb, MM, NKV * HD, Hh);
    gemm_xw_kernel<0><<<dim3((NKV * HD) / BN, MM / BM), 256, 0, stream>>>(
        xb, Wv + (size_t)l * NKV * HD * Hh, vb, MM, NKV * HD, Hh);
    rope_kernel<<<(MM * NH * 64 + 255) / 256, 256, 0, stream>>>(qb, NH, MM * NH * 64);
    rope_kernel<<<(MM * NKV * 64 + 255) / 256, 256, 0, stream>>>(kb, NKV, MM * NKV * 64);
    vtrans_kernel<<<(MM * NKV * HD + 255) / 256, 256, 0, stream>>>(vb, vtb);
    attn_kernel<<<dim3(Bb * NH, Ss / 64), 128, 64 * 1032 * 4, stream>>>(qb, kb, vtb, ctxb);
    gemm_xw_kernel<2><<<dim3(Hh / BN, MM / BM), 256, 0, stream>>>(
        ctxb, Wo + (size_t)l * Hh * NH * HD, h, MM, Hh, NH * HD);
    rmsnorm_kernel<<<MM, 256, 0, stream>>>(h, pnrm + (size_t)l * Hh, xb);
    gemm_xw_kernel<0><<<dim3(FF / BN, MM / BM), 256, 0, stream>>>(
        xb, Wg + (size_t)l * FF * Hh, gb, MM, FF, Hh);
    gemm_xw_kernel<0><<<dim3(FF / BN, MM / BM), 256, 0, stream>>>(
        xb, Wu + (size_t)l * FF * Hh, ub, MM, FF, Hh);
    silu_mul_kernel<<<(MM * FF + 255) / 256, 256, 0, stream>>>(gb, ub, fb, MM * FF);
    gemm_xw_kernel<2><<<dim3(Hh / BN, MM / BM), 256, 0, stream>>>(
        fb, Wd + (size_t)l * Hh * FF, h, MM, Hh, FF);
  }
  rmsnorm_kernel<<<MM, 256, 0, stream>>>(h, fnrm, xb);
  gemm_xw_kernel<1><<<dim3(Vv / BN, MM / BM), 256, 0, stream>>>(
      xb, lmh, out, MM, Vv, Hh);

  (void)in_sizes; (void)n_in; (void)out_size; (void)ws_size;
}
